// PFNet6_6408091206057
// MI455X (gfx1250) — compile-verified, matrix operand order checked by reading, not run on hardware
//
#include <hip/hip_runtime.h>
#include <hip/hip_bf16.h>

// ---------------------------------------------------------------------------
// PFNet6 forward for gfx1250 (MI455X, wave32, WMMA).
// All GEMMs run on v_wmma_f32_16x16x32_f16; activations staged in LDS as f16.
// ---------------------------------------------------------------------------

typedef __attribute__((ext_vector_type(16))) _Float16 v16h;
typedef __attribute__((ext_vector_type(8)))  float    v8f;

#define WMMA_F16(a, b, c) \
  __builtin_amdgcn_wmma_f32_16x16x32_f16(false, (a), false, (b), (short)0, (c), false, false)

// Explicit CDNA5 split-counter LDS fence (also blocks compiler reordering).
#define LDS_FENCE() asm volatile("s_wait_dscnt 0" ::: "memory")

union FragU { v16h v; unsigned u[8]; };

// Load a 16x32 f16 A-fragment from a row-major LDS tile.
// CDNA5 16-bit A layout: lane l (l<16: rows, K groups 0..7/16..23 ; l>=16 adds +8).
__device__ inline v16h load_afrag(const _Float16* t, int stride, int lane, int kt) {
  FragU f;
  int m = lane & 15, g = lane >> 4;
  const _Float16* row = t + m * stride + kt * 32 + g * 8;
#pragma unroll
  for (int j = 0; j < 8; ++j) {
    int k = ((j >> 2) << 4) + ((j & 3) << 1);   // 0,2,4,6,16,18,20,22
    f.u[j] = *(const unsigned*)(row + k);       // two consecutive halves
  }
  return f.v;
}

// One hidden MLP layer: LDS(in) -> WMMA -> fused BN + LeakyReLU(0.01) -> LDS(out)
template <int KT>
__device__ inline void mlp_layer(const _Float16* in, int sin, _Float16* out, int sout,
                                 const _Float16* __restrict__ wf, int NT,
                                 const float* __restrict__ sc, const float* __restrict__ sh,
                                 int lane) {
  LDS_FENCE();
  v16h a[KT];
#pragma unroll
  for (int kt = 0; kt < KT; ++kt) a[kt] = load_afrag(in, sin, lane, kt);
  int g = lane >> 4, nn = lane & 15;
  for (int nt = 0; nt < NT; ++nt) {
    v8f c = {};
#pragma unroll
    for (int kt = 0; kt < KT; ++kt) {
      v16h b = *(const v16h*)(wf + (((size_t)kt * NT + nt) * 32 + lane) * 16);
      c = WMMA_F16(a[kt], b, c);
    }
    int ncol = nt * 16 + nn;
    float s = sc[ncol], t = sh[ncol];
#pragma unroll
    for (int r = 0; r < 8; ++r) {
      float y = c[r] * s + t;
      y = (y >= 0.f) ? y : 0.01f * y;
      out[(r + g * 8) * sout + ncol] = (_Float16)y;
    }
  }
}

// Order-preserving float<->uint for atomicMax-based segment max.
__device__ inline unsigned fenc(float f) {
  unsigned u = __float_as_uint(f);
  return (u & 0x80000000u) ? ~u : (u | 0x80000000u);
}
__device__ inline float fdec(unsigned k) {
  unsigned u = (k & 0x80000000u) ? (k & 0x7fffffffu) : ~k;
  return __uint_as_float(u);
}

struct MlpW {
  const _Float16* w[5];   // 4 block weight fragments + final
  const float*    sc[4];
  const float*    sh[4];
  const float*    fb;     // final bias
};

// ---------------------------- prep kernels ---------------------------------

// Pack a KxN f32 weight into CDNA5 f16 B-fragment layout (zero padded).
__global__ void k_bfrag(const float* __restrict__ W, _Float16* __restrict__ f,
                        int K, int N, int KT, int NT) {
  int idx = blockIdx.x * blockDim.x + threadIdx.x;
  int tot = KT * NT * 512;
  if (idx >= tot) return;
  int e = idx & 15, lane = (idx >> 4) & 31, t = idx >> 9;
  int nt = t % NT, kt = t / NT;
  int k = kt * 32 + ((e >> 3) << 4) + ((lane >> 4) << 3) + (e & 7);
  int n = nt * 16 + (lane & 15);
  f[idx] = (_Float16)((k < K && n < N) ? W[(size_t)k * N + n] : 0.f);
}

__global__ void k_bn(const float* __restrict__ g, const float* __restrict__ be,
                     const float* __restrict__ me, const float* __restrict__ va,
                     float* __restrict__ sc, float* __restrict__ sh, int n) {
  int i = blockIdx.x * blockDim.x + threadIdx.x;
  if (i >= n) return;
  float s = g[i] * rsqrtf(va[i] + 1e-5f);
  sc[i] = s;
  sh[i] = be[i] - me[i] * s;
}

// ---------------------------- node MLP kernels -----------------------------

// input MLP: x(16) -> 4x [64 + BN + lrelu] -> linear 64 -> h0 (f16)
__global__ void k_inp(const float* __restrict__ x, MlpW p,
                      _Float16* __restrict__ h0, int n) {
  __shared__ _Float16 sm[8][2048];
  int wave = threadIdx.x >> 5, lane = threadIdx.x & 31;
  int row0 = (blockIdx.x * 8 + wave) * 16;
  if (row0 >= n) return;
  _Float16* A = sm[wave];
  _Float16* B = A + 1024;
  if (lane < 16) {
    int node = row0 + lane; if (node >= n) node = n - 1;
    const float* xr = x + (size_t)node * 16;
#pragma unroll
    for (int c = 0; c < 16; ++c) A[lane * 32 + c] = (_Float16)xr[c];
#pragma unroll
    for (int c = 16; c < 32; ++c) A[lane * 32 + c] = (_Float16)0.f;
  }
  mlp_layer<1>(A, 32, B, 64, p.w[0], 4, p.sc[0], p.sh[0], lane);
  mlp_layer<2>(B, 64, A, 64, p.w[1], 4, p.sc[1], p.sh[1], lane);
  mlp_layer<2>(A, 64, B, 64, p.w[2], 4, p.sc[2], p.sh[2], lane);
  mlp_layer<2>(B, 64, A, 64, p.w[3], 4, p.sc[3], p.sh[3], lane);
  LDS_FENCE();
  v16h a0 = load_afrag(A, 64, lane, 0), a1 = load_afrag(A, 64, lane, 1);
  int g = lane >> 4, nn = lane & 15;
  for (int nt = 0; nt < 4; ++nt) {
    v8f c = {};
    v16h b0 = *(const v16h*)(p.w[4] + (((size_t)(0 * 4 + nt)) * 32 + lane) * 16);
    v16h b1 = *(const v16h*)(p.w[4] + (((size_t)(1 * 4 + nt)) * 32 + lane) * 16);
    c = WMMA_F16(a0, b0, c);
    c = WMMA_F16(a1, b1, c);
    int ncol = nt * 16 + nn;
    float bb = p.fb[ncol];
#pragma unroll
    for (int r = 0; r < 8; ++r) {
      int node = row0 + r + g * 8;
      if (node < n) h0[(size_t)node * 64 + ncol] = (_Float16)(c[r] + bb);
    }
  }
}

// GAT linear projection hW = h @ W (no bias), f32 out for aggregation.
__global__ void k_gatw(const _Float16* __restrict__ h, const _Float16* __restrict__ wf,
                       float* __restrict__ hw, int n) {
  __shared__ _Float16 sm[8][1024];
  int wave = threadIdx.x >> 5, lane = threadIdx.x & 31;
  int row0 = (blockIdx.x * 8 + wave) * 16;
  if (row0 >= n) return;
  _Float16* A = sm[wave];
  if (lane < 16) {
    int node = row0 + lane; if (node >= n) node = n - 1;
    const uint4* s4 = (const uint4*)(h + (size_t)node * 64);
    uint4* d4 = (uint4*)(A + lane * 64);
    d4[0] = s4[0]; d4[1] = s4[1]; d4[2] = s4[2]; d4[3] = s4[3];
  }
  LDS_FENCE();
  v16h a0 = load_afrag(A, 64, lane, 0), a1 = load_afrag(A, 64, lane, 1);
  int g = lane >> 4, nn = lane & 15;
  for (int nt = 0; nt < 4; ++nt) {
    v8f c = {};
    v16h b0 = *(const v16h*)(wf + (((size_t)(0 * 4 + nt)) * 32 + lane) * 16);
    v16h b1 = *(const v16h*)(wf + (((size_t)(1 * 4 + nt)) * 32 + lane) * 16);
    c = WMMA_F16(a0, b0, c);
    c = WMMA_F16(a1, b1, c);
    int ncol = nt * 16 + nn;
#pragma unroll
    for (int r = 0; r < 8; ++r) {
      int node = row0 + r + g * 8;
      if (node < n) hw[(size_t)node * 64 + ncol] = c[r];
    }
  }
}

// ---------------------------- GAT softmax kernels --------------------------

__global__ void k_dots(const float* __restrict__ hw, const float* __restrict__ att_dst,
                       const float* __restrict__ att_src, float* __restrict__ asrc,
                       float* __restrict__ adst, float* __restrict__ slog,
                       unsigned* __restrict__ mkey, int n) {
  int i = blockIdx.x * blockDim.x + threadIdx.x;
  if (i >= n) return;
  const float* h = hw + (size_t)i * 64;
  float s = 0.f, d = 0.f;
#pragma unroll 8
  for (int f = 0; f < 64; ++f) { float v = h[f]; s += v * att_src[f]; d += v * att_dst[f]; }
  asrc[i] = s; adst[i] = d;
  float l = s + d;
  l = (l >= 0.f) ? l : 0.2f * l;     // self-loop logit (mask always 1)
  slog[i] = l;
  mkey[i] = fenc(l);
}

__global__ void k_edge_max(const float* __restrict__ asrc, const float* __restrict__ adst,
                           const int* __restrict__ src, const int* __restrict__ dst,
                           const float* __restrict__ ew, float* __restrict__ logit,
                           unsigned* __restrict__ mkey, int ne) {
  int e = blockIdx.x * blockDim.x + threadIdx.x;
  if (e >= ne) return;
  int s = src[e], d = dst[e];
  float l = asrc[s] + adst[d];
  l = (l >= 0.f) ? l : 0.2f * l;
  logit[e] = l;
  bool m = ew ? (ew[e] > 0.5f) : true;
  if (m) atomicMax(&mkey[d], fenc(l));
}

__global__ void k_denom_init(const unsigned* __restrict__ mkey, const float* __restrict__ slog,
                             float* __restrict__ mval, float* __restrict__ denom, int n) {
  int i = blockIdx.x * blockDim.x + threadIdx.x;
  if (i >= n) return;
  float mv = fdec(mkey[i]);
  mval[i] = mv;
  denom[i] = expf(slog[i] - mv);     // self-loop term
}

__global__ void k_edge_sum(const float* __restrict__ logit, const float* __restrict__ mval,
                           float* __restrict__ denom, const int* __restrict__ dst,
                           const float* __restrict__ ew, float* __restrict__ eval, int ne) {
  int e = blockIdx.x * blockDim.x + threadIdx.x;
  if (e >= ne) return;
  bool m = ew ? (ew[e] > 0.5f) : true;
  float ev = 0.f;
  if (m) {
    int d = dst[e];
    ev = expf(logit[e] - mval[d]);
    atomicAdd(&denom[d], ev);
  }
  eval[e] = ev;
}

__global__ void k_self_agg(const float* __restrict__ hw, const float* __restrict__ slog,
                           const float* __restrict__ mval, const float* __restrict__ denom,
                           float* __restrict__ agg, int n) {
  int idx = blockIdx.x * blockDim.x + threadIdx.x;
  if (idx >= n * 64) return;
  int i = idx >> 6;
  float al = expf(slog[i] - mval[i]) / fmaxf(denom[i], 1e-16f);
  agg[idx] = hw[idx] * al;
}

__global__ void k_edge_agg(const float* __restrict__ hw, const float* __restrict__ eval,
                           const float* __restrict__ denom, const int* __restrict__ src,
                           const int* __restrict__ dst, float* __restrict__ agg, int ne) {
  int idx = blockIdx.x * blockDim.x + threadIdx.x;
  if (idx >= ne * 64) return;
  int e = idx >> 6, f = idx & 63;
  float ev = eval[e];
  if (ev > 0.f) {
    int s = src[e], d = dst[e];
    float al = ev / fmaxf(denom[d], 1e-16f);
    atomicAdd(&agg[(size_t)d * 64 + f], hw[(size_t)s * 64 + f] * al);
  }
}

__global__ void k_gat_finish(const float* __restrict__ agg, const float* __restrict__ bias,
                             _Float16* __restrict__ hout, int n) {
  int idx = blockIdx.x * blockDim.x + threadIdx.x;
  if (idx >= n * 64) return;
  float y = agg[idx] + bias[idx & 63];
  y = (y >= 0.f) ? y : 0.01f * y;    // outer lrelu
  hout[idx] = (_Float16)y;
}

// ---------------------------- edgenet kernel -------------------------------

// Per-edge MLP: [h1[src](64) | h1[dst](64) | w(1) | pad] -> 4x64 -> 1, sigmoid.
__global__ void k_edgenet(const _Float16* __restrict__ h1, const int* __restrict__ src,
                          const int* __restrict__ dst, const float* __restrict__ eattr,
                          MlpW p, float* __restrict__ out_e, int ne) {
  __shared__ _Float16 sm[4][2560 + 1024];
  int wave = threadIdx.x >> 5, lane = threadIdx.x & 31;
  int row0 = (blockIdx.x * 4 + wave) * 16;
  if (row0 >= ne) return;
  _Float16* A = sm[wave];
  _Float16* B = A + 2560;
  if (lane < 16) {
    int e = row0 + lane; if (e >= ne) e = ne - 1;
    int s = src[e], d = dst[e];
    const uint4* s4 = (const uint4*)(h1 + (size_t)s * 64);
    const uint4* d4 = (const uint4*)(h1 + (size_t)d * 64);
    uint4* a4 = (uint4*)(A + lane * 160);
    a4[0] = s4[0]; a4[1] = s4[1]; a4[2] = s4[2]; a4[3] = s4[3];
    a4[4] = d4[0]; a4[5] = d4[1]; a4[6] = d4[2]; a4[7] = d4[3];
    unsigned* aw = (unsigned*)(A + lane * 160 + 128);
    union { _Float16 h[2]; unsigned u; } wv;
    wv.h[0] = (_Float16)eattr[e]; wv.h[1] = (_Float16)0.f;
    aw[0] = wv.u;
#pragma unroll
    for (int j = 1; j < 16; ++j) aw[j] = 0u;
  }
  mlp_layer<5>(A, 160, B, 64, p.w[0], 4, p.sc[0], p.sh[0], lane);
  mlp_layer<2>(B,  64, A, 64, p.w[1], 4, p.sc[1], p.sh[1], lane);
  mlp_layer<2>(A,  64, B, 64, p.w[2], 4, p.sc[2], p.sh[2], lane);
  mlp_layer<2>(B,  64, A, 64, p.w[3], 4, p.sc[3], p.sh[3], lane);
  LDS_FENCE();
  v16h a0 = load_afrag(A, 64, lane, 0), a1 = load_afrag(A, 64, lane, 1);
  v8f c = {};
  v16h b0 = *(const v16h*)(p.w[4] + ((size_t)0 * 32 + lane) * 16);
  v16h b1 = *(const v16h*)(p.w[4] + ((size_t)1 * 32 + lane) * 16);
  c = WMMA_F16(a0, b0, c);
  c = WMMA_F16(a1, b1, c);
  if ((lane & 15) == 0) {
    int g = lane >> 4;
    float bb = p.fb[0];
#pragma unroll
    for (int r = 0; r < 8; ++r) {
      int e = row0 + r + g * 8;
      if (e < ne) {
        float v = c[r] + bb;
        out_e[e] = 1.f / (1.f + expf(-v));
      }
    }
  }
}

// ---------------------------- nn1 / nn2 kernels ----------------------------

__global__ void k_nn1(const _Float16* __restrict__ h2, MlpW p,
                      float* __restrict__ out_ids, _Float16* __restrict__ cand16, int n) {
  __shared__ _Float16 sm[8][2048];
  int wave = threadIdx.x >> 5, lane = threadIdx.x & 31;
  int row0 = (blockIdx.x * 8 + wave) * 16;
  if (row0 >= n) return;
  _Float16* A = sm[wave];
  _Float16* B = A + 1024;
  if (lane < 16) {
    int node = row0 + lane; if (node >= n) node = n - 1;
    const uint4* s4 = (const uint4*)(h2 + (size_t)node * 64);
    uint4* d4 = (uint4*)(A + lane * 64);
    d4[0] = s4[0]; d4[1] = s4[1]; d4[2] = s4[2]; d4[3] = s4[3];
  }
  mlp_layer<2>(A, 64, B, 64, p.w[0], 4, p.sc[0], p.sh[0], lane);
  mlp_layer<2>(B, 64, A, 64, p.w[1], 4, p.sc[1], p.sh[1], lane);
  mlp_layer<2>(A, 64, B, 64, p.w[2], 4, p.sc[2], p.sh[2], lane);
  mlp_layer<2>(B, 64, A, 64, p.w[3], 4, p.sc[3], p.sh[3], lane);
  LDS_FENCE();
  v16h a0 = load_afrag(A, 64, lane, 0), a1 = load_afrag(A, 64, lane, 1);
  v8f c = {};
  v16h b0 = *(const v16h*)(p.w[4] + ((size_t)0 * 32 + lane) * 16);
  v16h b1 = *(const v16h*)(p.w[4] + ((size_t)1 * 32 + lane) * 16);
  c = WMMA_F16(a0, b0, c);
  c = WMMA_F16(a1, b1, c);
  int g = lane >> 4, nn = lane & 15;
  if (nn < 6) {
    float bb = p.fb[nn];
#pragma unroll
    for (int r = 0; r < 8; ++r) {
      int node = row0 + r + g * 8;
      if (node < n) {
        float y = c[r] + bb;
        out_ids[(size_t)node * 6 + nn] = y;
        cand16[(size_t)node * 6 + nn] = (_Float16)y;
      }
    }
  }
}

__global__ void k_nn2(const _Float16* __restrict__ h2, const _Float16* __restrict__ cand16,
                      const float* __restrict__ x, MlpW p, float* __restrict__ out_p4, int n) {
  __shared__ _Float16 sm[8][1536 + 1024];
  int wave = threadIdx.x >> 5, lane = threadIdx.x & 31;
  int row0 = (blockIdx.x * 8 + wave) * 16;
  if (row0 >= n) return;
  _Float16* A = sm[wave];
  _Float16* B = A + 1536;
  if (lane < 16) {
    int node = row0 + lane; if (node >= n) node = n - 1;
    unsigned* ar = (unsigned*)(A + lane * 96);
    const unsigned* cu = (const unsigned*)(cand16 + (size_t)node * 6);
    ar[0] = cu[0]; ar[1] = cu[1]; ar[2] = cu[2];               // cols 0..5
    const unsigned* hu = (const unsigned*)(h2 + (size_t)node * 64);
#pragma unroll
    for (int j = 0; j < 32; ++j) ar[3 + j] = hu[j];            // cols 6..69
#pragma unroll
    for (int j = 35; j < 48; ++j) ar[j] = 0u;                  // cols 70..95
  }
  mlp_layer<3>(A, 96, B, 64, p.w[0], 4, p.sc[0], p.sh[0], lane);
  mlp_layer<2>(B, 64, A, 64, p.w[1], 4, p.sc[1], p.sh[1], lane);
  mlp_layer<2>(A, 64, B, 64, p.w[2], 4, p.sc[2], p.sh[2], lane);
  mlp_layer<2>(B, 64, A, 64, p.w[3], 4, p.sc[3], p.sh[3], lane);
  LDS_FENCE();
  v16h a0 = load_afrag(A, 64, lane, 0), a1 = load_afrag(A, 64, lane, 1);
  v8f c = {};
  v16h b0 = *(const v16h*)(p.w[4] + ((size_t)0 * 32 + lane) * 16);
  v16h b1 = *(const v16h*)(p.w[4] + ((size_t)1 * 32 + lane) * 16);
  c = WMMA_F16(a0, b0, c);
  c = WMMA_F16(a1, b1, c);
  int g = lane >> 4, nn = lane & 15;
  if (nn < 4) {
    float bb = p.fb[nn];
#pragma unroll
    for (int r = 0; r < 8; ++r) {
      int node = row0 + r + g * 8;
      if (node < n) {
        float y = c[r] + bb + x[(size_t)node * 16 + 12 + nn];  // x[:,12:16] + delta
        out_p4[(size_t)node * 4 + nn] = y;
      }
    }
  }
}

// ---------------------------- host side ------------------------------------

static void* ws_alloc(char* ws, size_t& off, size_t bytes) {
  off = (off + 255) & ~(size_t)255;
  void* p = ws + off;
  off += bytes;
  return p;
}

// Leaf order within one _mlp: blocks[0..3] x (W,b,beta,gamma,mean,var), final (W,b)
static void setup_mlp(void* const* d_in, int base, int K0, int Nf,
                      MlpW& mw, char* ws, size_t& off, hipStream_t st) {
  int KT0 = (K0 + 31) / 32;
  int NTf = (Nf + 15) / 16;
  int KT[5] = {KT0, 2, 2, 2, 2};
  int NT[5] = {4, 4, 4, 4, NTf};
  int K[5]  = {K0, 64, 64, 64, 64};
  int N[5]  = {64, 64, 64, 64, Nf};
  for (int i = 0; i < 5; ++i) {
    int tot = KT[i] * NT[i] * 512;
    _Float16* f = (_Float16*)ws_alloc(ws, off, (size_t)tot * 2);
    const float* W = (const float*)d_in[(i < 4) ? (base + 6 * i) : (base + 24)];
    k_bfrag<<<(tot + 255) / 256, 256, 0, st>>>(W, f, K[i], N[i], KT[i], NT[i]);
    mw.w[i] = f;
  }
  for (int i = 0; i < 4; ++i) {
    float* sc = (float*)ws_alloc(ws, off, 64 * 4);
    float* sh = (float*)ws_alloc(ws, off, 64 * 4);
    int b = base + 6 * i;
    k_bn<<<1, 64, 0, st>>>((const float*)d_in[b + 3], (const float*)d_in[b + 2],
                           (const float*)d_in[b + 4], (const float*)d_in[b + 5], sc, sh, 64);
    mw.sc[i] = sc; mw.sh[i] = sh;
  }
  mw.fb = (const float*)d_in[base + 25];
}

static void run_gat(const _Float16* hin, const _Float16* wfrag,
                    const float* att_dst, const float* att_src, const float* bias,
                    const int* src, const int* dst, const float* ew,
                    float* hw, float* asrc, float* adst, float* slog, float* mval,
                    float* denom, unsigned* mkey, float* logit, float* eval,
                    float* agg, _Float16* hout, int n, int ne, hipStream_t st) {
  int ntiles = (n + 15) / 16;
  k_gatw<<<(ntiles + 7) / 8, 256, 0, st>>>(hin, wfrag, hw, n);
  k_dots<<<(n + 255) / 256, 256, 0, st>>>(hw, att_dst, att_src, asrc, adst, slog, mkey, n);
  k_edge_max<<<(ne + 255) / 256, 256, 0, st>>>(asrc, adst, src, dst, ew, logit, mkey, ne);
  k_denom_init<<<(n + 255) / 256, 256, 0, st>>>(mkey, slog, mval, denom, n);
  k_edge_sum<<<(ne + 255) / 256, 256, 0, st>>>(logit, mval, denom, dst, ew, eval, ne);
  k_self_agg<<<(n * 64 + 255) / 256, 256, 0, st>>>(hw, slog, mval, denom, agg, n);
  k_edge_agg<<<(ne / 4 * 64 + 255) / 256 * 4, 256, 0, st>>>(hw, eval, denom, src, dst, agg, ne);
  k_gat_finish<<<(n * 64 + 255) / 256, 256, 0, st>>>(agg, bias, hout, n);
}

extern "C" void kernel_launch(void* const* d_in, const int* in_sizes, int n_in,
                              void* d_out, int out_size, void* d_ws, size_t ws_size,
                              hipStream_t stream) {
  (void)in_sizes; (void)n_in; (void)out_size; (void)ws_size;
  const int N = 100000, NE = 1600000;

  const float* x     = (const float*)d_in[0];
  const int*   ei    = (const int*)d_in[1];
  const float* eattr = (const float*)d_in[2];
  const int* src = ei;
  const int* dst = ei + NE;

  float* out     = (float*)d_out;
  float* out_e   = out;                         // edge_weight2 (NE)
  float* out_ids = out + NE;                    // cand_ids (N*6)
  float* out_p4  = out + NE + (size_t)N * 6;    // cand_p4 (N*4)

  char* ws = (char*)d_ws;
  size_t off = 0;

  // Param leaf base indices (params dict keys sorted: edgenet,gat1,gat2,inp,nn1,nn2)
  const int B_EDGENET = 3, B_GAT1 = 29, B_GAT2 = 33, B_INP = 37, B_NN1 = 63, B_NN2 = 89;

  MlpW p_en, p_inp, p_n1, p_n2;
  setup_mlp(d_in, B_EDGENET, 129, 1, p_en, ws, off, stream);
  setup_mlp(d_in, B_INP,      16, 64, p_inp, ws, off, stream);
  setup_mlp(d_in, B_NN1,      64,  6, p_n1, ws, off, stream);
  setup_mlp(d_in, B_NN2,      70,  4, p_n2, ws, off, stream);

  _Float16* g1w = (_Float16*)ws_alloc(ws, off, 2 * 4 * 512 * 2);
  k_bfrag<<<(2 * 4 * 512 + 255) / 256, 256, 0, stream>>>((const float*)d_in[B_GAT1], g1w, 64, 64, 2, 4);
  _Float16* g2w = (_Float16*)ws_alloc(ws, off, 2 * 4 * 512 * 2);
  k_bfrag<<<(2 * 4 * 512 + 255) / 256, 256, 0, stream>>>((const float*)d_in[B_GAT2], g2w, 64, 64, 2, 4);

  _Float16* h0 = (_Float16*)ws_alloc(ws, off, (size_t)N * 64 * 2);
  _Float16* h1 = (_Float16*)ws_alloc(ws, off, (size_t)N * 64 * 2);
  _Float16* h2 = h0;  // h0 dead after gat1 projection -> reuse
  _Float16* cand16 = (_Float16*)ws_alloc(ws, off, (size_t)N * 6 * 2);
  float* hw    = (float*)ws_alloc(ws, off, (size_t)N * 64 * 4);
  float* agg   = (float*)ws_alloc(ws, off, (size_t)N * 64 * 4);
  float* asrc  = (float*)ws_alloc(ws, off, (size_t)N * 4);
  float* adst  = (float*)ws_alloc(ws, off, (size_t)N * 4);
  float* slog  = (float*)ws_alloc(ws, off, (size_t)N * 4);
  float* mval  = (float*)ws_alloc(ws, off, (size_t)N * 4);
  float* denom = (float*)ws_alloc(ws, off, (size_t)N * 4);
  unsigned* mkey = (unsigned*)ws_alloc(ws, off, (size_t)N * 4);
  float* logit = (float*)ws_alloc(ws, off, (size_t)NE * 4);
  float* eval  = (float*)ws_alloc(ws, off, (size_t)NE * 4);

  int ntiles = (N + 15) / 16;
  int etiles = (NE + 15) / 16;

  // 1) input MLP
  k_inp<<<(ntiles + 7) / 8, 256, 0, stream>>>(x, p_inp, h0, N);

  // 2) GAT1 (mask = all ones) -> h1
  run_gat(h0, g1w, (const float*)d_in[B_GAT1 + 1], (const float*)d_in[B_GAT1 + 2],
          (const float*)d_in[B_GAT1 + 3], src, dst, nullptr,
          hw, asrc, adst, slog, mval, denom, mkey, logit, eval, agg, h1, N, NE, stream);

  // 3) edgenet -> edge_weight2 (sigmoid) straight into d_out
  k_edgenet<<<(etiles + 3) / 4, 128, 0, stream>>>(h1, src, dst, eattr, p_en, out_e, NE);

  // 4) GAT2 (mask = edge_weight2 > 0.5) -> h2
  run_gat(h1, g2w, (const float*)d_in[B_GAT2 + 1], (const float*)d_in[B_GAT2 + 2],
          (const float*)d_in[B_GAT2 + 3], src, dst, out_e,
          hw, asrc, adst, slog, mval, denom, mkey, logit, eval, agg, h2, N, NE, stream);

  // 5) heads
  k_nn1<<<(ntiles + 7) / 8, 256, 0, stream>>>(h2, p_n1, out_ids, cand16, N);
  k_nn2<<<(ntiles + 7) / 8, 256, 0, stream>>>(h2, cand16, x, p_n2, out_p4, N);
}
